// PaGNNAggregation_51101520888248
// MI455X (gfx1250) — matrix operand order repaired and checked by reading, not compiled, not run on hardware
//
#include <hip/hip_runtime.h>

#define HD 128
#define SA 20   // LDS A row stride (dwords): 16 data + 4 pad, 16B-multiple
#define SB 36   // LDS B row stride (generic): 32 data + 4 pad
#define SBL 68  // LDS B row stride (LSTM): 64 data + 4 pad

typedef __attribute__((ext_vector_type(16))) __bf16 v16bf;
typedef __attribute__((ext_vector_type(8)))  float  v8f;

union Frag16 { v16bf v; unsigned int u[8]; };

__device__ inline unsigned int pack2_bf16(float a, float b) {
  unsigned int ua = __float_as_uint(a); ua = (ua + 0x7FFFu + ((ua >> 16) & 1u)) >> 16;
  unsigned int ub = __float_as_uint(b); ub = (ub + 0x7FFFu + ((ub >> 16) & 1u)) >> 16;
  return (ua & 0xFFFFu) | (ub << 16);
}
__device__ inline unsigned short f2bf(float a) {
  unsigned int u = __float_as_uint(a);
  u = (u + 0x7FFFu + ((u >> 16) & 1u)) >> 16;
  return (unsigned short)u;
}
__device__ inline float sigmoidf_(float x) { return 1.f / (1.f + __expf(-x)); }

// float atomic-max via sign-split int atomics (init memory to -FLT_MAX)
__device__ inline void atomicMaxF(float* addr, float v) {
  if (v >= 0.f) atomicMax((int*)addr, __float_as_int(v));
  else          atomicMin((unsigned int*)addr, __float_as_uint(v));
}

// ---- CDNA5 async global->LDS copy (ASYNCcnt-tracked, no VGPR data movement)
// lds_byte_addr: LDS offset (low 32 bits of the generic shared address)
__device__ inline void async_b128(unsigned lds_byte_addr, const void* gaddr) {
  asm volatile("global_load_async_to_lds_b128 %0, %1, off"
               :: "v"(lds_byte_addr), "v"(gaddr)
               : "memory");
}
__device__ inline void wait_async_all() {
  asm volatile("s_wait_asynccnt 0x0" ::: "memory");
}

// ---------------------------------------------------------------- utilities
// f32 pairs -> packed bf16 dwords (RNE)
__global__ void pack_bf16_kernel(const float* __restrict__ s,
                                 unsigned int* __restrict__ d, int n2) {
  int i = blockIdx.x * 256 + threadIdx.x;
  if (i < n2) {
    float2 v = ((const float2*)s)[i];
    d[i] = pack2_bf16(v.x, v.y);
  }
}

__global__ void fill_f32_kernel(float* __restrict__ p, float v, size_t n) {
  size_t i = (size_t)blockIdx.x * 256 + threadIdx.x;
  if (i < n) p[i] = v;
}

// ---------------------------------------------------------- generic WMMA GEMM
// C[M,ldc] = concat(A0|A1)[M,K](bf16) * B[K,ldb](bf16) (+bias).
// Block tile 128x64, 8 waves as 4x2, each wave 32x32 (2x2 frags), K-step 32.
__global__ __launch_bounds__(256)
void gemm_bf16_kernel(const unsigned short* __restrict__ A0,
                      const unsigned short* __restrict__ A1,
                      int K0, int K,
                      const unsigned short* __restrict__ Bw, int ldb,
                      const float* __restrict__ bias,
                      void* __restrict__ Cout, int ldc, int M, int outBf16) {
  __shared__ unsigned int ldsA[128 * SA];
  __shared__ unsigned int ldsB[32 * SB];
  const int tid  = threadIdx.x;
  const int row0 = blockIdx.x * 128;
  const int col0 = blockIdx.y * 64;
  const int w = tid >> 5, lane = tid & 31;
  const int wr = w >> 1, wc = w & 1;
  const int half = lane >> 4, hl = lane & 15;
  const int K1 = K - K0;

  v8f acc[2][2];
#pragma unroll
  for (int mi = 0; mi < 2; mi++)
#pragma unroll
    for (int ni = 0; ni < 2; ni++)
#pragma unroll
      for (int r = 0; r < 8; r++) acc[mi][ni][r] = 0.f;

  const unsigned int* B32 = (const unsigned int*)Bw;
  const int ldbD = ldb >> 1;
  for (int kk = 0; kk < K; kk += 32) {
    // select concat segment uniformly per K-tile (tiles never straddle K0)
    const unsigned int* Abase = (kk < K0) ? (const unsigned int*)A0 : (const unsigned int*)A1;
    const int ldA = ((kk < K0) ? K0 : K1) >> 1;
    const int kd  = ((kk < K0) ? kk : kk - K0) >> 1;
    // A tile: 128 rows x 16 dwords = 512 async b128, 2 per thread.
    // OOB rows clamped (their results are discarded by the guarded epilogue).
#pragma unroll
    for (int i = 0; i < 2; i++) {
      int q = tid + i * 256;          // 0..511
      int r = q >> 2;                 // row 0..127
      int dc = (q & 3) * 4;           // dword col 0,4,8,12
      int gr = row0 + r;
      if (gr >= M) gr = M - 1;
      async_b128((unsigned)(size_t)&ldsA[r * SA + dc],
                 Abase + (size_t)gr * ldA + kd + dc);
    }
    // B tile: 32 rows x 32 dwords = 256 async b128, 1 per thread
    {
      int r = tid >> 3;               // 0..31
      int nc = (tid & 7) * 4;         // 0..28
      async_b128((unsigned)(size_t)&ldsB[r * SB + nc],
                 B32 + (size_t)(kk + r) * ldbD + (col0 >> 1) + nc);
    }
    wait_async_all();
    __syncthreads();

    Frag16 af[2], bf[2];
#pragma unroll
    for (int mi = 0; mi < 2; mi++) {
      const unsigned int* base = &ldsA[(wr * 32 + mi * 16 + hl) * SA];
      *(uint4*)&af[mi].u[0] = *(const uint4*)&base[half * 4];       // K 0..7 / 8..15
      *(uint4*)&af[mi].u[4] = *(const uint4*)&base[8 + half * 4];   // K 16..23 / 24..31
    }
#pragma unroll
    for (int ni = 0; ni < 2; ni++) {
      const unsigned short* base = (const unsigned short*)ldsB + wc * 32 + ni * 16 + hl;  // n = hl, k table order
#pragma unroll
      for (int j = 0; j < 8; j++) {
        int k = (j < 4) ? (8 * half + 2 * j) : (16 + 8 * half + 2 * (j - 4));
        bf[ni].u[j] = (unsigned int)base[k * (2 * SB)] | ((unsigned int)base[(k + 1) * (2 * SB)] << 16);
      }
    }
#pragma unroll
    for (int mi = 0; mi < 2; mi++)
#pragma unroll
      for (int ni = 0; ni < 2; ni++)
        acc[mi][ni] = __builtin_amdgcn_wmma_f32_16x16x32_bf16(
            false, af[mi].v, false, bf[ni].v, (short)0, acc[mi][ni], false, false);
    __syncthreads();
  }

  float* Cf = (float*)Cout;
  unsigned short* Cb = (unsigned short*)Cout;
#pragma unroll
  for (int mi = 0; mi < 2; mi++) {
    int mrow = row0 + wr * 32 + mi * 16 + half * 8;
#pragma unroll
    for (int ni = 0; ni < 2; ni++) {
      int ncol = col0 + wc * 32 + ni * 16 + hl;
      float bv = bias ? bias[ncol] : 0.f;
#pragma unroll
      for (int r = 0; r < 8; r++) {
        int gr = mrow + r;
        if (gr < M) {
          float val = acc[mi][ni][r] + bv;
          if (outBf16) Cb[(size_t)gr * ldc + ncol] = f2bf(val);
          else         Cf[(size_t)gr * ldc + ncol] = val;
        }
      }
    }
  }
}

// ------------------------------------------------------------- LSTM step
union SmemLSTM {
  struct { unsigned int A[64 * SA]; unsigned int B[32 * SBL]; } tiles;
  float g[64 * 132];
};

__device__ inline void lstm_gemm_phase(unsigned int* ldsA, unsigned int* ldsB,
                                       v8f acc[4],
                                       const unsigned short* __restrict__ A0,
                                       const unsigned short* __restrict__ A1,
                                       int K0, int Ktot,
                                       const unsigned int* __restrict__ B32, int jb,
                                       int row0, int Nn, int tid) {
  const int w = tid >> 5, lane = tid & 31;
  const int wr = w & 3, wc = w >> 2;
  const int half = lane >> 4, hl = lane & 15;
  const int K1 = Ktot - K0;
  for (int kk = 0; kk < Ktot; kk += 32) {
    const unsigned int* Abase = (kk < K0) ? (const unsigned int*)A0 : (const unsigned int*)A1;
    const int ldA = ((kk < K0) ? K0 : K1) >> 1;
    const int kd  = ((kk < K0) ? kk : kk - K0) >> 1;
    {   // A tile: 64 rows x 16 dwords = 256 async b128, 1 per thread
      int r = tid >> 2;               // 0..63
      int dc = (tid & 3) * 4;
      int gr = row0 + r;
      if (gr >= Nn) gr = Nn - 1;
      async_b128((unsigned)(size_t)&ldsA[r * SA + dc],
                 Abase + (size_t)gr * ldA + kd + dc);
    }
    // B tile: 32 rows x 64 virtual dwords = 512 async b128, 2 per thread
    // virtual cols 0-31:i 32-63:f 64-95:g 96-127:o ; strip jb selects 32 h-cols
#pragma unroll
    for (int i = 0; i < 2; i++) {
      int q = tid + i * 256;          // 0..511
      int r = q >> 4;                 // k row 0..31
      int vcd = (q & 15) * 4;         // virtual dword col 0,4,..,60
      int gate = vcd >> 4;
      int dwcol = gate * 64 + jb * 16 + (vcd & 15);
      async_b128((unsigned)(size_t)&ldsB[r * SBL + vcd],
                 &B32[(size_t)(kk + r) * 256 + dwcol]);   // ldb=512 -> 256 dwords
    }
    wait_async_all();
    __syncthreads();

    Frag16 af, bfr[4];
    {
      const unsigned int* base = &ldsA[(wr * 16 + hl) * SA];
      *(uint4*)&af.u[0] = *(const uint4*)&base[half * 4];
      *(uint4*)&af.u[4] = *(const uint4*)&base[8 + half * 4];
    }
#pragma unroll
    for (int ni = 0; ni < 4; ni++) {
      const unsigned short* base = (const unsigned short*)ldsB + wc * 64 + ni * 16 + hl;
#pragma unroll
      for (int j = 0; j < 8; j++) {
        int k = (j < 4) ? (8 * half + 2 * j) : (16 + 8 * half + 2 * (j - 4));
        bfr[ni].u[j] = (unsigned int)base[k * (2 * SBL)] | ((unsigned int)base[(k + 1) * (2 * SBL)] << 16);
      }
    }
#pragma unroll
    for (int ni = 0; ni < 4; ni++)
      acc[ni] = __builtin_amdgcn_wmma_f32_16x16x32_bf16(
          false, af.v, false, bfr[ni].v, (short)0, acc[ni], false, false);
    __syncthreads();
  }
}

// gates = [htmp|xs] @ Wih + hin @ Whh + b ; LSTM elementwise -> cbuf(f32), hout(bf16)
// grid: (ceil(N/64), 4 strips of 32 h-cols), block 256
__global__ __launch_bounds__(256)
void lstm_step_kernel(const unsigned short* __restrict__ htmp,
                      const unsigned short* __restrict__ xs,
                      const unsigned short* __restrict__ hin,
                      unsigned int* __restrict__ hout32,
                      float* __restrict__ cbuf,
                      const unsigned short* __restrict__ Wih_bf,
                      const unsigned short* __restrict__ Whh_bf,
                      const float* __restrict__ bvec, int Nn) {
  __shared__ SmemLSTM sm;
  const int tid = threadIdx.x;
  const int row0 = blockIdx.x * 64;
  const int jb = blockIdx.y;

  v8f acc[4];
#pragma unroll
  for (int ni = 0; ni < 4; ni++)
#pragma unroll
    for (int r = 0; r < 8; r++) acc[ni][r] = 0.f;

  lstm_gemm_phase(sm.tiles.A, sm.tiles.B, acc, htmp, xs, HD, 2 * HD,
                  (const unsigned int*)Wih_bf, jb, row0, Nn, tid);
  lstm_gemm_phase(sm.tiles.A, sm.tiles.B, acc, hin, hin, HD, HD,
                  (const unsigned int*)Whh_bf, jb, row0, Nn, tid);

  // stage gates (+bias) into LDS (aliases tiles; phases end with a barrier)
  const int w = tid >> 5, lane = tid & 31;
  const int wr = w & 3, wc = w >> 2;
  const int half = lane >> 4, hl = lane & 15;
#pragma unroll
  for (int ni = 0; ni < 4; ni++) {
    int vcol = wc * 64 + ni * 16 + hl;
    int acol = (vcol >> 5) * HD + jb * 32 + (vcol & 31);
    float bv = bvec[acol];
#pragma unroll
    for (int r = 0; r < 8; r++)
      sm.g[(wr * 16 + half * 8 + r) * 132 + vcol] = acc[ni][r] + bv;
  }
  __syncthreads();

#pragma unroll
  for (int i = 0; i < 4; i++) {
    int d = tid + i * 256;            // 64 rows x 16 paired h-cols
    int r = d >> 4, dc = d & 15;
    int node = row0 + r;
    if (node < Nn) {
      int lc = dc * 2;
      float gi0 = sm.g[r * 132 + lc],      gi1 = sm.g[r * 132 + lc + 1];
      float gf0 = sm.g[r * 132 + 32 + lc], gf1 = sm.g[r * 132 + 33 + lc];
      float gg0 = sm.g[r * 132 + 64 + lc], gg1 = sm.g[r * 132 + 65 + lc];
      float go0 = sm.g[r * 132 + 96 + lc], go1 = sm.g[r * 132 + 97 + lc];
      size_t base = (size_t)node * HD + jb * 32 + lc;
      float2 c2 = *(float2*)&cbuf[base];
      float cn0 = sigmoidf_(gf0) * c2.x + sigmoidf_(gi0) * tanhf(gg0);
      float cn1 = sigmoidf_(gf1) * c2.y + sigmoidf_(gi1) * tanhf(gg1);
      float2 cw; cw.x = cn0; cw.y = cn1;
      *(float2*)&cbuf[base] = cw;
      float h0 = sigmoidf_(go0) * tanhf(cn0);
      float h1 = sigmoidf_(go1) * tanhf(cn1);
      hout32[(size_t)node * (HD / 2) + jb * 16 + dc] = pack2_bf16(h0, h1);
    }
  }
}

// ------------------------------------------------------------- GAT edge stage
__global__ __launch_bounds__(256)
void scores_kernel(const float* __restrict__ xw, const float* __restrict__ asrc,
                   const float* __restrict__ adst, float* __restrict__ ssrc,
                   float* __restrict__ sdst, int Nn) {
  int w = threadIdx.x >> 5, lane = threadIdx.x & 31;
  int n = blockIdx.x * 8 + w;
  if (n >= Nn) return;
  float4 xv = ((const float4*)(xw + (size_t)n * HD))[lane];
  float4 a4 = ((const float4*)asrc)[lane];
  float4 d4 = ((const float4*)adst)[lane];
  float s0 = xv.x * a4.x + xv.y * a4.y + xv.z * a4.z + xv.w * a4.w;
  float s1 = xv.x * d4.x + xv.y * d4.y + xv.z * d4.z + xv.w * d4.w;
#pragma unroll
  for (int off = 16; off; off >>= 1) {
    s0 += __shfl_xor(s0, off, 32);
    s1 += __shfl_xor(s1, off, 32);
  }
  if (lane == 0) { ssrc[n] = s0; sdst[n] = s1; }
}

__global__ void edge_score_kernel(const int* __restrict__ src, const int* __restrict__ dst,
                                  const float* __restrict__ ssrc, const float* __restrict__ sdst,
                                  float* __restrict__ es, float* __restrict__ m, int Ne) {
  int e = blockIdx.x * 256 + threadIdx.x;
  if (e >= Ne) return;
  float s = ssrc[src[e]] + sdst[dst[e]];
  s = s > 0.f ? s : 0.2f * s;                // leaky_relu(0.2)
  es[e] = s;
  atomicMaxF(&m[dst[e]], s);
}

__global__ void edge_exp_kernel(const int* __restrict__ dst, const float* __restrict__ m,
                                float* __restrict__ es, float* __restrict__ z, int Ne) {
  int e = blockIdx.x * 256 + threadIdx.x;
  if (e >= Ne) return;
  int d = dst[e];
  float ex = __expf(es[e] - m[d]);
  es[e] = ex;
  atomicAdd(&z[d], ex);
}

__global__ __launch_bounds__(256)
void edge_agg_kernel(const int* __restrict__ src, const int* __restrict__ dst,
                     const float* __restrict__ es, const float* __restrict__ z,
                     const float* __restrict__ xw, float* __restrict__ agg, int Ne) {
  int w = threadIdx.x >> 5, lane = threadIdx.x & 31;
  int e = blockIdx.x * 8 + w;
  if (e >= Ne) return;
  int s = src[e], d = dst[e];
  float alpha = es[e] / z[d];
  float4 v = ((const float4*)(xw + (size_t)s * HD))[lane];
  float* ar = agg + (size_t)d * HD + lane * 4;
  atomicAdd(ar + 0, alpha * v.x);
  atomicAdd(ar + 1, alpha * v.y);
  atomicAdd(ar + 2, alpha * v.z);
  atomicAdd(ar + 3, alpha * v.w);
}

// htmp_bf16 = tanh(agg + b) ; 4 elems per thread
__global__ void bias_tanh_bf16_kernel(const float* __restrict__ agg,
                                      const float* __restrict__ b,
                                      unsigned int* __restrict__ out32, int n4) {
  int i = blockIdx.x * 256 + threadIdx.x;
  if (i >= n4) return;
  float4 v = ((const float4*)agg)[i];
  int cb = (i * 4) & (HD - 1);
  float4 bb = *(const float4*)&b[cb];
  float t0 = tanhf(v.x + bb.x), t1 = tanhf(v.y + bb.y);
  float t2 = tanhf(v.z + bb.z), t3 = tanhf(v.w + bb.w);
  out32[i * 2]     = pack2_bf16(t0, t1);
  out32[i * 2 + 1] = pack2_bf16(t2, t3);
}

__global__ void gather_rows_kernel(const unsigned int* __restrict__ xsrc32,
                                   const int* __restrict__ roots,
                                   const int* __restrict__ aggflag,
                                   unsigned int* __restrict__ rows32, int Bn) {
  int b = blockIdx.x, t = threadIdx.x;       // blockDim = 64 (dwords per bf16 row)
  if (b >= Bn) return;
  int sel = (aggflag[0] != 0) ? 0 : 1;
  int node = roots[2 * b + sel];
  rows32[(size_t)b * (HD / 2) + t] = xsrc32[(size_t)node * (HD / 2) + t];
}

// ---------------------------------------------------------------- driver
extern "C" void kernel_launch(void* const* d_in, const int* in_sizes, int n_in,
                              void* d_out, int out_size, void* d_ws, size_t ws_size,
                              hipStream_t stream) {
  (void)n_in; (void)out_size; (void)ws_size;
  const float* node_feat = (const float*)d_in[0];
  const float* send_embd = (const float*)d_in[1];
  const int*   edge      = (const int*)d_in[2];
  const int*   roots     = (const int*)d_in[3];
  const int*   aggflag   = (const int*)d_in[4];
  const float* lin1W = (const float*)d_in[5];
  const float* lin1b = (const float*)d_in[6];
  const float* gatW  = (const float*)d_in[7];
  const float* gatAs = (const float*)d_in[8];
  const float* gatAd = (const float*)d_in[9];
  const float* gatB  = (const float*)d_in[10];
  const float* Wih   = (const float*)d_in[11];
  const float* Whh   = (const float*)d_in[12];
  const float* lstmB = (const float*)d_in[13];
  const float* lin2W = (const float*)d_in[14];
  const float* lin2b = (const float*)d_in[15];

  const int Nn   = in_sizes[0] / HD;
  const int Ne   = in_sizes[2] / 2;
  const int Bp   = in_sizes[3] / 2;
  const int Hops = in_sizes[7] / (HD * HD);
  const int* srcIdx = edge;
  const int* dstIdx = edge + Ne;
  const size_t NH = (size_t)Nn * HD;

  size_t off = 0;
  auto carve = [&](size_t bytes) -> void* {
    void* p = (void*)((char*)d_ws + off);
    off += (bytes + 255) & ~(size_t)255;
    return p;
  };
  // f32 scratch
  float* xw   = (float*)carve(NH * 4);
  float* agg  = (float*)carve(NH * 4);
  float* cbuf = (float*)carve(NH * 4);
  float* ssrc = (float*)carve((size_t)Nn * 4);
  float* sdst = (float*)carve((size_t)Nn * 4);
  float* mbuf = (float*)carve((size_t)Nn * 4);
  float* zbuf = (float*)carve((size_t)Nn * 4);
  float* exb  = (float*)carve((size_t)Ne * 4);
  // bf16 activations
  unsigned short* nf_bf   = (unsigned short*)carve(NH * 2);
  unsigned short* se_bf   = (unsigned short*)carve(NH * 2);
  unsigned short* x_bf    = (unsigned short*)carve(NH * 2);
  unsigned short* htmp_bf = (unsigned short*)carve(NH * 2);
  unsigned short* h0_bf   = (unsigned short*)carve(NH * 2);
  unsigned short* h1_bf   = (unsigned short*)carve(NH * 2);
  unsigned short* rows_bf = (unsigned short*)carve((size_t)Bp * HD * 2);
  // bf16 weights
  unsigned short* lin1Wb = (unsigned short*)carve((size_t)2 * HD * HD * 2);
  unsigned short* gatWb  = (unsigned short*)carve((size_t)Hops * HD * HD * 2);
  unsigned short* Wihb   = (unsigned short*)carve((size_t)Hops * 2 * HD * 4 * HD * 2);
  unsigned short* Whhb   = (unsigned short*)carve((size_t)Hops * HD * 4 * HD * 2);
  unsigned short* lin2Wb = (unsigned short*)carve((size_t)HD * HD * 2);

  auto cvt = [&](const float* s, unsigned short* dgt, size_t n) {
    int n2 = (int)(n / 2);
    pack_bf16_kernel<<<dim3((n2 + 255) / 256), dim3(256), 0, stream>>>(
        s, (unsigned int*)dgt, n2);
  };
  cvt(lin1W, lin1Wb, (size_t)2 * HD * HD);
  cvt(gatW,  gatWb,  (size_t)Hops * HD * HD);
  cvt(Wih,   Wihb,   (size_t)Hops * 2 * HD * 4 * HD);
  cvt(Whh,   Whhb,   (size_t)Hops * HD * 4 * HD);
  cvt(lin2W, lin2Wb, (size_t)HD * HD);
  cvt(node_feat, nf_bf, NH);
  cvt(send_embd, se_bf, NH);

  // x = [node_feat | send_embd] @ lin1_W + lin1_b  -> bf16
  gemm_bf16_kernel<<<dim3((Nn + 127) / 128, 2), dim3(256), 0, stream>>>(
      nf_bf, se_bf, HD, 2 * HD, lin1Wb, HD, lin1b, x_bf, HD, Nn, 1);

  fill_f32_kernel<<<dim3((unsigned)((NH / 2 + 255) / 256)), dim3(256), 0, stream>>>(
      (float*)h0_bf, 0.f, NH / 2);
  fill_f32_kernel<<<dim3((unsigned)((NH + 255) / 256)), dim3(256), 0, stream>>>(cbuf, 0.f, NH);

  unsigned short* hin = h0_bf;
  unsigned short* hout = h1_bf;
  const unsigned short* xs = x_bf;
  for (int hop = 0; hop < Hops; hop++) {
    // breadth: xw = x @ gat_W[hop] (f32 out for edge stage)
    gemm_bf16_kernel<<<dim3((Nn + 127) / 128, 2), dim3(256), 0, stream>>>(
        x_bf, x_bf, HD, HD, gatWb + (size_t)hop * HD * HD, HD, nullptr, xw, HD, Nn, 0);
    scores_kernel<<<dim3((Nn + 7) / 8), dim3(256), 0, stream>>>(
        xw, gatAs + hop * HD, gatAd + hop * HD, ssrc, sdst, Nn);
    fill_f32_kernel<<<dim3((Nn + 255) / 256), dim3(256), 0, stream>>>(mbuf, -3.4028235e38f, (size_t)Nn);
    fill_f32_kernel<<<dim3((Nn + 255) / 256), dim3(256), 0, stream>>>(zbuf, 0.f, (size_t)Nn);
    fill_f32_kernel<<<dim3((unsigned)((NH + 255) / 256)), dim3(256), 0, stream>>>(agg, 0.f, NH);
    edge_score_kernel<<<dim3((Ne + 255) / 256), dim3(256), 0, stream>>>(srcIdx, dstIdx, ssrc, sdst, exb, mbuf, Ne);
    edge_exp_kernel<<<dim3((Ne + 255) / 256), dim3(256), 0, stream>>>(dstIdx, mbuf, exb, zbuf, Ne);
    edge_agg_kernel<<<dim3((Ne + 7) / 8), dim3(256), 0, stream>>>(srcIdx, dstIdx, exb, zbuf, xw, agg, Ne);
    bias_tanh_bf16_kernel<<<dim3((unsigned)((NH / 4 + 255) / 256)), dim3(256), 0, stream>>>(
        agg, gatB + hop * HD, (unsigned int*)htmp_bf, (int)(NH / 4));
    // depth: fused LSTM step (ping-pong bf16 h)
    lstm_step_kernel<<<dim3((Nn + 63) / 64, 4), dim3(256), 0, stream>>>(
        htmp_bf, xs, hin, (unsigned int*)hout, cbuf,
        Wihb + (size_t)hop * 2 * HD * 4 * HD,
        Whhb + (size_t)hop * HD * 4 * HD,
        lstmB + hop * 4 * HD, Nn);
    xs = hout;
    unsigned short* t = hin; hin = hout; hout = t;
  }

  // gather root rows (bf16), then out = rows @ lin2_W + lin2_b (f32 -> d_out)
  gather_rows_kernel<<<dim3(Bp), dim3(HD / 2), 0, stream>>>(
      (const unsigned int*)xs, roots, aggflag, (unsigned int*)rows_bf, Bp);
  gemm_bf16_kernel<<<dim3((Bp + 127) / 128, 2), dim3(256), 0, stream>>>(
      rows_bf, rows_bf, HD, HD, lin2Wb, HD, lin2b, d_out, HD, Bp, 0);
}
